// DGRUCell_25391846654692
// MI455X (gfx1250) — compile-verified, matrix operand order checked by reading, not run on hardware
//
#include <hip/hip_runtime.h>
#include <hip/hip_bf16.h>

// ---------------------------------------------------------------------------
// DGRU cell, MI455X (gfx1250) — WMMA f16 path with async global->LDS staging.
//   k0: f32 -> f16 weight convert (Wg, Wu)
//   k1: LN(concat(x,h)) -> f16 activations
//   k2: WMMA GEMM  gates(f16) = A @ Wg^T + bg      (EPI=0)
//   k3: sigmoid/softmax gates, hpart/z2 (f16), LN2 -> f16 activations
//   k4: WMMA GEMM  out = hpart + tanh(A2 @ Wu^T + bu) * z2   (EPI=1, fused)
// GEMM staging uses GLOBAL_LOAD_ASYNC_TO_LDS_B128 + s_wait_asynccnt
// (ASYNCcnt path) with double-buffered LDS, BK=64 (16 WMMA per barrier/wave).
// ---------------------------------------------------------------------------

typedef __attribute__((ext_vector_type(16))) _Float16 v16h;
typedef __attribute__((ext_vector_type(8)))  _Float16 v8h;
typedef __attribute__((ext_vector_type(8)))  float    v8f;

#define DIMD   512
#define TWO_D  1024
#define FIVE_D 2560
#define BATCHN 32768
#define LNEPS  1e-5f

// ----------------------------- CDNA5 async helpers -------------------------

// 16B per lane, global -> LDS, tracked by ASYNCcnt (CDNA5 ISA).
__device__ __forceinline__ void async_b128(unsigned lds_off, const void* gptr) {
    asm volatile("global_load_async_to_lds_b128 %0, %1, off"
                 :: "v"(lds_off), "v"(gptr)
                 : "memory");
}

__device__ __forceinline__ void wait_async0() {
#if __has_builtin(__builtin_amdgcn_s_wait_asynccnt)
    __builtin_amdgcn_s_wait_asynccnt(0);
#else
    asm volatile("s_wait_asynccnt 0x0" ::: "memory");
#endif
}

// ----------------------------- helpers -------------------------------------

__device__ __forceinline__ void block_reduce2(float& s, float& q, float* sm) {
    // 256 threads = 8 wave32s; reduce sum and sum-of-squares block-wide.
    #pragma unroll
    for (int o = 16; o > 0; o >>= 1) {
        s += __shfl_xor(s, o, 32);
        q += __shfl_xor(q, o, 32);
    }
    const int lane = threadIdx.x & 31;
    const int wid  = threadIdx.x >> 5;
    if (lane == 0) { sm[wid] = s; sm[8 + wid] = q; }
    __syncthreads();
    if (wid == 0) {
        float ws = (lane < 8) ? sm[lane]     : 0.f;
        float wq = (lane < 8) ? sm[8 + lane] : 0.f;
        #pragma unroll
        for (int o = 4; o > 0; o >>= 1) {
            ws += __shfl_xor(ws, o, 32);
            wq += __shfl_xor(wq, o, 32);
        }
        if (lane == 0) { sm[0] = ws; sm[1] = wq; }
    }
    __syncthreads();
    s = sm[0]; q = sm[1];
}

// Load one 16x32 (f16) WMMA operand fragment from LDS.
// ISA layout (CDNA5 ISA): lane&15 = row (M or N), two contiguous 8-half
// chunks at K = (lane>>4)*8 and K = (lane>>4)*8 + 16.
__device__ __forceinline__ v16h load_frag(const _Float16* base, int ld, int lane) {
    const int r  = lane & 15;
    const int k0 = (lane >> 4) << 3;
    const _Float16* p = base + r * ld + k0;
    v8h lo = *(const v8h*)(p);
    v8h hi = *(const v8h*)(p + 16);
    v16h f;
    #pragma unroll
    for (int i = 0; i < 8; ++i) { f[i] = lo[i]; f[i + 8] = hi[i]; }
    return f;
}

// ----------------------------- kernels -------------------------------------

__global__ __launch_bounds__(256)
void f32_to_f16_kernel(const float* __restrict__ src, _Float16* __restrict__ dst, int n) {
    int i = blockIdx.x * 256 + threadIdx.x;
    if (i < n) dst[i] = (_Float16)src[i];
}

// LN1: out[row, 0:2d] = LN(concat(x,h)) * ln_w + ln_b, times dropout mask, as f16
__global__ __launch_bounds__(256)
void ln_concat_kernel(const float* __restrict__ x, const float* __restrict__ h,
                      const float* __restrict__ w, const float* __restrict__ b,
                      const float* __restrict__ mask, _Float16* __restrict__ out) {
    __shared__ float sm[16];
    const int row = blockIdx.x;
    const int c0  = threadIdx.x * 4;
    const float* xr = x + (size_t)row * DIMD;
    const float* hr = h + (size_t)row * DIMD;
    float v[4], s = 0.f, q = 0.f;
    #pragma unroll
    for (int i = 0; i < 4; ++i) {
        int c = c0 + i;
        float t = (c < DIMD) ? xr[c] : hr[c - DIMD];
        v[i] = t; s += t; q += t * t;
    }
    block_reduce2(s, q, sm);
    const float mean = s * (1.f / TWO_D);
    const float rstd = rsqrtf(q * (1.f / TWO_D) - mean * mean + LNEPS);
    _Float16* o = out + (size_t)row * TWO_D + c0;
    #pragma unroll
    for (int i = 0; i < 4; ++i) {
        int c = c0 + i;
        o[i] = (_Float16)(((v[i] - mean) * rstd * w[c] + b[c]) * mask[c]);
    }
}

// Gates(f16) -> rx/rh/softmax(z); hpart = x*z0 + h*z1 (f16); z2 (f16);
// LN2(concat(x*rx, h*rh)) -> f16
__global__ __launch_bounds__(256)
void gate_ln2_kernel(const float* __restrict__ x, const float* __restrict__ h,
                     const _Float16* __restrict__ gates,
                     const float* __restrict__ w, const float* __restrict__ b,
                     const float* __restrict__ mask,
                     _Float16* __restrict__ inp2,
                     _Float16* __restrict__ hpart, _Float16* __restrict__ z2out) {
    __shared__ float sm[16];
    const int row = blockIdx.x;
    const int tid = threadIdx.x;
    const _Float16* g = gates + (size_t)row * FIVE_D;
    const float* xr = x + (size_t)row * DIMD;
    const float* hr = h + (size_t)row * DIMD;
    float t1[2], t2[2], s = 0.f, q = 0.f;
    #pragma unroll
    for (int u = 0; u < 2; ++u) {
        const int j = tid + u * 256;
        const float xv = xr[j], hv = hr[j];
        const float rx = 1.f / (1.f + __expf(-(float)g[j]));
        const float rh = 1.f / (1.f + __expf(-(float)g[j + DIMD]));
        const float za = (float)g[j + 2 * DIMD];
        const float zb = (float)g[j + 3 * DIMD];
        const float zc = (float)g[j + 4 * DIMD];
        const float mx = fmaxf(za, fmaxf(zb, zc));
        const float ea = __expf(za - mx), eb = __expf(zb - mx), ec = __expf(zc - mx);
        const float inv = 1.f / (ea + eb + ec);
        hpart[(size_t)row * DIMD + j] = (_Float16)(xv * ea * inv + hv * eb * inv);
        z2out[(size_t)row * DIMD + j] = (_Float16)(ec * inv);
        t1[u] = xv * rx; t2[u] = hv * rh;
        s += t1[u] + t2[u];
        q += t1[u] * t1[u] + t2[u] * t2[u];
    }
    block_reduce2(s, q, sm);
    const float mean = s * (1.f / TWO_D);
    const float rstd = rsqrtf(q * (1.f / TWO_D) - mean * mean + LNEPS);
    _Float16* o = inp2 + (size_t)row * TWO_D;
    #pragma unroll
    for (int u = 0; u < 2; ++u) {
        const int j = tid + u * 256;
        o[j]        = (_Float16)(((t1[u] - mean) * rstd * w[j] + b[j]) * mask[j]);
        o[j + DIMD] = (_Float16)(((t2[u] - mean) * rstd * w[j + DIMD] + b[j + DIMD]) * mask[j + DIMD]);
    }
}

// WMMA GEMM: C[M,N] = A[M,K](f16) @ W[N,K](f16)^T + bias.
// Both operands are K-major, matching the ISA's per-lane fragment layout.
// Staging: async global->LDS (ASYNCcnt), double-buffered, BK=64.
// EPI=0: store f16 gates.  EPI=1: out(f32) = hpart + tanh(acc+bias)*z2 (fused).
template <int EPI>
__global__ __launch_bounds__(256)
void gemm_wmma_kernel(const _Float16* __restrict__ A,
                      const _Float16* __restrict__ W,
                      const float* __restrict__ bias,
                      void* __restrict__ Cout,
                      const _Float16* __restrict__ hpart,
                      const _Float16* __restrict__ z2,
                      int M, int N, int K) {
    constexpr int BM = 128, BN = 128, BK = 64;
    constexpr int LDT = BK + 8;  // 144B row stride: 16B aligned, spreads banks
    __shared__ __align__(16) _Float16 sA[2][BM * LDT];
    __shared__ __align__(16) _Float16 sB[2][BN * LDT];

    const int tid  = threadIdx.x;
    const int lane = tid & 31;
    const int wid  = tid >> 5;
    const int wm   = wid & 1;   // 2 wave-rows  -> 64 rows each
    const int wn   = wid >> 1;  // 4 wave-cols  -> 32 cols each
    const int bm   = blockIdx.y * BM;
    const int bn   = blockIdx.x * BN;

    // Async staging: tile = 128 rows x 8 chunks(8 halves) = 1024 chunks;
    // 4 chunks per thread per tile, 16B each, straight into LDS.
    auto stage = [&](int kt, int buf) {
        const int ka = kt * BK;
        #pragma unroll
        for (int t = 0; t < 4; ++t) {
            const int c   = tid + t * 256;
            const int row = c >> 3;
            const int kc  = (c & 7) * 8;
            async_b128((unsigned)(uintptr_t)&sA[buf][row * LDT + kc],
                       A + (size_t)(bm + row) * K + ka + kc);
            async_b128((unsigned)(uintptr_t)&sB[buf][row * LDT + kc],
                       W + (size_t)(bn + row) * K + ka + kc);
        }
    };

    v8f acc[4][2] = {};

    stage(0, 0);
    wait_async0();
    __syncthreads();

    const int nk = K / BK;
    for (int kt = 0; kt < nk; ++kt) {
        const int buf = kt & 1;
        if (kt + 1 < nk) stage(kt + 1, buf ^ 1);

        #pragma unroll
        for (int kk = 0; kk < 2; ++kk) {
            const int ko = kk * 32;
            v16h af[4], bf[2];
            #pragma unroll
            for (int i = 0; i < 4; ++i)
                af[i] = load_frag(&sA[buf][(wm * 64 + i * 16) * LDT + ko], LDT, lane);
            #pragma unroll
            for (int j = 0; j < 2; ++j)
                bf[j] = load_frag(&sB[buf][(wn * 32 + j * 16) * LDT + ko], LDT, lane);

            #pragma unroll
            for (int i = 0; i < 4; ++i)
                #pragma unroll
                for (int j = 0; j < 2; ++j)
                    acc[i][j] = __builtin_amdgcn_wmma_f32_16x16x32_f16(
                        false, af[i], false, bf[j], (short)0, acc[i][j], false, false);
        }

        wait_async0();   // own async writes into buf^1 done
        __syncthreads(); // everyone's done
    }

    // Epilogue. C/D layout: VGPR r of lane l holds (M = r + 8*(l>>4), N = l&15).
    const int mm = bm + wm * 64;
    const int nn = bn + wn * 32;
    #pragma unroll
    for (int i = 0; i < 4; ++i) {
        #pragma unroll
        for (int j = 0; j < 2; ++j) {
            const int col  = nn + j * 16 + (lane & 15);
            const int rowb = mm + i * 16 + ((lane >> 4) << 3);
            const float bv = bias[col];
            #pragma unroll
            for (int r = 0; r < 8; ++r) {
                const size_t idx = (size_t)(rowb + r) * N + col;
                const float v = acc[i][j][r] + bv;
                if (EPI == 0) {
                    ((_Float16*)Cout)[idx] = (_Float16)v;
                } else {
                    ((float*)Cout)[idx] =
                        (float)hpart[idx] + tanhf(v) * (float)z2[idx];
                }
            }
        }
    }
}

// ----------------------------- launcher ------------------------------------

extern "C" void kernel_launch(void* const* d_in, const int* in_sizes, int n_in,
                              void* d_out, int out_size, void* d_ws, size_t ws_size,
                              hipStream_t stream) {
    const float* x     = (const float*)d_in[0];
    const float* h     = (const float*)d_in[1];
    const float* Wg    = (const float*)d_in[2];
    const float* bg    = (const float*)d_in[3];
    const float* Wu    = (const float*)d_in[4];
    const float* bu    = (const float*)d_in[5];
    const float* ln_w  = (const float*)d_in[6];
    const float* ln_b  = (const float*)d_in[7];
    const float* ln2_w = (const float*)d_in[8];
    const float* ln2_b = (const float*)d_in[9];
    const float* mask  = (const float*)d_in[10];
    float* out = (float*)d_out;

    char* wsb = (char*)d_ws;
    size_t off = 0;
    auto wsalloc = [&](size_t bytes) -> char* {
        char* p = wsb + off;
        off += (bytes + 255) & ~(size_t)255;
        return p;
    };
    _Float16* inpA  = (_Float16*)wsalloc((size_t)BATCHN * TWO_D * 2);
    _Float16* Wg16  = (_Float16*)wsalloc((size_t)FIVE_D * TWO_D * 2);
    _Float16* Wu16  = (_Float16*)wsalloc((size_t)DIMD * TWO_D * 2);
    _Float16* gates = (_Float16*)wsalloc((size_t)BATCHN * FIVE_D * 2);
    _Float16* inp2  = (_Float16*)wsalloc((size_t)BATCHN * TWO_D * 2);
    _Float16* hpart = (_Float16*)wsalloc((size_t)BATCHN * DIMD * 2);
    _Float16* z2    = (_Float16*)wsalloc((size_t)BATCHN * DIMD * 2);

    f32_to_f16_kernel<<<(FIVE_D * TWO_D + 255) / 256, 256, 0, stream>>>(Wg, Wg16, FIVE_D * TWO_D);
    f32_to_f16_kernel<<<(DIMD * TWO_D + 255) / 256, 256, 0, stream>>>(Wu, Wu16, DIMD * TWO_D);

    ln_concat_kernel<<<BATCHN, 256, 0, stream>>>(x, h, ln_w, ln_b, mask, inpA);

    gemm_wmma_kernel<0><<<dim3(FIVE_D / 128, BATCHN / 128), 256, 0, stream>>>(
        inpA, Wg16, bg, gates, nullptr, nullptr, BATCHN, FIVE_D, TWO_D);

    gate_ln2_kernel<<<BATCHN, 256, 0, stream>>>(x, h, gates, ln2_w, ln2_b, mask,
                                                inp2, hpart, z2);

    gemm_wmma_kernel<1><<<dim3(DIMD / 128, BATCHN / 128), 256, 0, stream>>>(
        inp2, Wu16, bu, out, hpart, z2, BATCHN, DIMD, TWO_D);
}
